// FlatQuantQwen2MLP_78434692759806
// MI455X (gfx1250) — 8-bit matrix operand order repaired and checked by reading, compiled, not run on hardware
//
#include <hip/hip_runtime.h>
#include <cstdint>
#include <cstddef>

typedef int v8i __attribute__((ext_vector_type(8)));
typedef int v4i_vec __attribute__((vector_size(4 * sizeof(int))));

#define H_DIM   3584
#define I_DIM   18944
#define UG_DL   56
#define UG_DR   64
#define DN_DL   128
#define DN_DR   148

#define AS1 __attribute__((address_space(1)))
#define AS3 __attribute__((address_space(3)))

#if __has_builtin(__builtin_amdgcn_global_load_async_to_lds_b128)
#define HAVE_ASYNC_LDS 1
#else
#define HAVE_ASYNC_LDS 0
#endif

__device__ __forceinline__ void stage16(const int8_t* __restrict__ g,
                                        int8_t* __restrict__ l) {
#if HAVE_ASYNC_LDS
    __builtin_amdgcn_global_load_async_to_lds_b128(
        (AS1 v4i_vec*)g, (AS3 v4i_vec*)l, 0, 0);
#else
    *(int4*)l = *(const int4*)g;
#endif
}

__device__ __forceinline__ void stage_wait() {
#if HAVE_ASYNC_LDS
#if __has_builtin(__builtin_amdgcn_s_wait_asynccnt)
    __builtin_amdgcn_s_wait_asynccnt(0);
#else
    asm volatile("s_wait_asynccnt 0x0" ::: "memory");
#endif
#endif
}

__global__ __launch_bounds__(256) void gauss_inv_t(
    const float* __restrict__ A0, const float* __restrict__ A1,
    const float* __restrict__ A2, const float* __restrict__ A3,
    float* __restrict__ out0, float* __restrict__ out1,
    float* __restrict__ out2, float* __restrict__ out3,
    float* __restrict__ scratch)
{
    const int Ns[4] = {UG_DL, UG_DR, DN_DL, DN_DR};
    int b = blockIdx.x;
    const float* A = (b == 0) ? A0 : (b == 1) ? A1 : (b == 2) ? A2 : A3;
    float* out     = (b == 0) ? out0 : (b == 1) ? out1 : (b == 2) ? out2 : out3;
    int N = Ns[b];
    int W = 2 * N;
    float* aug = scratch + (size_t)b * (148 * 296);
    int tid = threadIdx.x;

    for (int i = tid; i < N * W; i += blockDim.x) {
        int r = i / W, c = i - r * W;
        aug[i] = (c < N) ? A[r * N + c] : ((c - N) == r ? 1.0f : 0.0f);
    }
    __syncthreads();

    __shared__ float fac[160];
    for (int k = 0; k < N; ++k) {
        float pin = 1.0f / aug[k * W + k];
        __syncthreads();
        for (int j = tid; j < W; j += blockDim.x) aug[k * W + j] *= pin;
        __syncthreads();
        for (int i = tid; i < N; i += blockDim.x) fac[i] = aug[i * W + k];
        __syncthreads();
        for (int idx = tid; idx < N * W; idx += blockDim.x) {
            int r = idx / W;
            if (r != k) aug[idx] -= fac[r] * aug[k * W + (idx - r * W)];
        }
        __syncthreads();
    }
    for (int i = tid; i < N * N; i += blockDim.x) {
        int a = i / N, c = i - a * N;
        out[i] = aug[(size_t)c * W + N + a];
    }
}

extern __shared__ float kq_lds[];

__global__ __launch_bounds__(256) void kron_quant(
    const float* __restrict__ src,
    const float* __restrict__ Leff,
    const float* __restrict__ Reff,
    const float* __restrict__ diag,
    int dl, int dr, int recip,
    int8_t* __restrict__ qout, float* __restrict__ sout)
{
    int row = blockIdx.x;
    int n = dl * dr;
    float* xr  = kq_lds;
    float* tmp = kq_lds + n;
    const float* srow = src + (size_t)row * n;

    for (int i = threadIdx.x; i < n; i += blockDim.x) {
        float v = srow[i];
        float dg = diag[i];
        v = recip ? (v / dg) : (v * dg);
        xr[i] = v;
    }
    __syncthreads();

    for (int i = threadIdx.x; i < n; i += blockDim.x) {
        int a = i / dr, d = i - a * dr;
        const float* xa = xr + a * dr;
        float acc = 0.0f;
        for (int b = 0; b < dr; ++b) acc = fmaf(xa[b], Reff[b * dr + d], acc);
        tmp[i] = acc;
    }
    __syncthreads();

    float amax = 0.0f;
    for (int i = threadIdx.x; i < n; i += blockDim.x) {
        int c = i / dr, d = i - c * dr;
        float acc = 0.0f;
        for (int a = 0; a < dl; ++a) acc = fmaf(Leff[a * dl + c], tmp[a * dr + d], acc);
        xr[i] = acc;
        amax = fmaxf(amax, fabsf(acc));
    }

    __shared__ float red[256];
    red[threadIdx.x] = amax;
    __syncthreads();
    for (int s = 128; s > 0; s >>= 1) {
        if (threadIdx.x < (unsigned)s) red[threadIdx.x] = fmaxf(red[threadIdx.x], red[threadIdx.x + s]);
        __syncthreads();
    }
    float scale = fmaxf(red[0] / 7.0f, 1e-8f);
    float rs = 1.0f / scale;
    int8_t* qrow = qout + (size_t)row * n;
    for (int i = threadIdx.x; i < n; i += blockDim.x) {
        float q = rintf(xr[i] * rs);
        q = fminf(fmaxf(q, -8.0f), 7.0f);
        qrow[i] = (int8_t)q;
    }
    if (threadIdx.x == 0) sout[row] = scale;
}

__device__ __forceinline__ v8i frag_a(const int8_t* tile, int lane) {
    int mrow = lane & 15;
    int half = lane >> 4;
    v8i f;
#pragma unroll
    for (int v = 0; v < 8; ++v) {
        int ks = ((v >> 1) << 4) + ((v & 1) << 2) + (half << 3);
        f[v] = *(const int*)(tile + mrow * 64 + ks);
    }
    return f;
}
__device__ __forceinline__ v8i frag_b(const int8_t* tile, int lane) {
    int nrow = lane & 15;
    int half = lane >> 4;
    v8i f;
#pragma unroll
    for (int v = 0; v < 8; ++v) {
        int ks = ((v >> 1) << 4) + ((v & 1) << 2) + (half << 3);
        f[v] = *(const int*)(tile + nrow * 64 + ks);
    }
    return f;
}

#define ZERO8 {0,0,0,0,0,0,0,0}

__global__ __launch_bounds__(256) void gemm_swiglu(
    const int8_t* __restrict__ qx, const float* __restrict__ sx,
    const int8_t* __restrict__ qg, const float* __restrict__ sg,
    const int8_t* __restrict__ qu, const float* __restrict__ su,
    float* __restrict__ xact, int T, int K, int N)
{
    __shared__ __align__(16) int8_t As[128][64];
    __shared__ __align__(16) int8_t Bg[64][64];
    __shared__ __align__(16) int8_t Bu[64][64];

    int m0 = blockIdx.y * 128;
    int n0 = blockIdx.x * 64;
    int tid  = threadIdx.x;
    int wave = tid >> 5, lane = tid & 31;
    int mw = (wave & 3) * 32;
    int nw = (wave >> 2) * 32;

    v8i accg[2][2] = {{ZERO8, ZERO8}, {ZERO8, ZERO8}};
    v8i accu[2][2] = {{ZERO8, ZERO8}, {ZERO8, ZERO8}};

    int arow = tid >> 1;
    int acol = (tid & 1) * 32;
    int brow = (tid & 127) >> 1;
    int bcol = (tid & 1) * 32;

    const int8_t* gA  = &qx[(size_t)(m0 + arow) * K + acol];
    const int8_t* gBg = &qg[(size_t)(n0 + brow) * K + bcol];
    const int8_t* gBu = &qu[(size_t)(n0 + brow) * K + bcol];

    for (int k0 = 0; k0 < K; k0 += 64) {
        __syncthreads();
        stage16(gA + k0,      &As[arow][acol]);
        stage16(gA + k0 + 16, &As[arow][acol + 16]);
        if (tid < 128) {
            stage16(gBg + k0,      &Bg[brow][bcol]);
            stage16(gBg + k0 + 16, &Bg[brow][bcol + 16]);
        } else {
            stage16(gBu + k0,      &Bu[brow][bcol]);
            stage16(gBu + k0 + 16, &Bu[brow][bcol + 16]);
        }
        if (k0 + 64 < K) {
            __builtin_prefetch(gA + k0 + 64, 0, 3);
            __builtin_prefetch((tid < 128 ? gBg : gBu) + k0 + 64, 0, 3);
        }
        stage_wait();
        __syncthreads();

        v8i a0 = frag_a(&As[mw][0], lane);
        v8i a1 = frag_a(&As[mw + 16][0], lane);
        v8i g0 = frag_b(&Bg[nw][0], lane);
        v8i g1 = frag_b(&Bg[nw + 16][0], lane);
        v8i u0 = frag_b(&Bu[nw][0], lane);
        v8i u1 = frag_b(&Bu[nw + 16][0], lane);
        accg[0][0] = __builtin_amdgcn_wmma_i32_16x16x64_iu8(true, a0, true, g0, accg[0][0], false, false);
        accg[0][1] = __builtin_amdgcn_wmma_i32_16x16x64_iu8(true, a0, true, g1, accg[0][1], false, false);
        accg[1][0] = __builtin_amdgcn_wmma_i32_16x16x64_iu8(true, a1, true, g0, accg[1][0], false, false);
        accg[1][1] = __builtin_amdgcn_wmma_i32_16x16x64_iu8(true, a1, true, g1, accg[1][1], false, false);
        accu[0][0] = __builtin_amdgcn_wmma_i32_16x16x64_iu8(true, a0, true, u0, accu[0][0], false, false);
        accu[0][1] = __builtin_amdgcn_wmma_i32_16x16x64_iu8(true, a0, true, u1, accu[0][1], false, false);
        accu[1][0] = __builtin_amdgcn_wmma_i32_16x16x64_iu8(true, a1, true, u0, accu[1][0], false, false);
        accu[1][1] = __builtin_amdgcn_wmma_i32_16x16x64_iu8(true, a1, true, u1, accu[1][1], false, false);
    }

    int half = lane >> 4;
#pragma unroll
    for (int j = 0; j < 2; ++j) {
        int ocol = n0 + nw + j * 16 + (lane & 15);
        float sgo = sg[ocol], suo = su[ocol];
#pragma unroll
        for (int i = 0; i < 2; ++i) {
#pragma unroll
            for (int r = 0; r < 8; ++r) {
                int trow = m0 + mw + i * 16 + r + half * 8;
                float s = sx[trow];
                float g = (float)accg[i][j][r] * s * sgo;
                float u = (float)accu[i][j][r] * s * suo;
                xact[(size_t)trow * N + ocol] = (g / (1.0f + __expf(-g))) * u;
            }
        }
    }
}

__global__ __launch_bounds__(256) void gemm_down(
    const int8_t* __restrict__ qx, const float* __restrict__ sx,
    const int8_t* __restrict__ qw, const float* __restrict__ sw,
    float* __restrict__ out, int T, int K, int N)
{
    __shared__ __align__(16) int8_t As[128][64];
    __shared__ __align__(16) int8_t Bs[64][64];

    int m0 = blockIdx.y * 128;
    int n0 = blockIdx.x * 64;
    int tid  = threadIdx.x;
    int wave = tid >> 5, lane = tid & 31;
    int mw = (wave & 3) * 32;
    int nw = (wave >> 2) * 32;

    v8i acc[2][2] = {{ZERO8, ZERO8}, {ZERO8, ZERO8}};

    int arow = tid >> 1;
    int acol = (tid & 1) * 32;
    int brow = (tid & 127) >> 1;
    int bcol = (tid & 1) * 32;

    const int8_t* gA = &qx[(size_t)(m0 + arow) * K + acol];
    const int8_t* gB = &qw[(size_t)(n0 + brow) * K + bcol];

    for (int k0 = 0; k0 < K; k0 += 64) {
        __syncthreads();
        stage16(gA + k0,      &As[arow][acol]);
        stage16(gA + k0 + 16, &As[arow][acol + 16]);
        if (tid < 128) {
            stage16(gB + k0,      &Bs[brow][bcol]);
            stage16(gB + k0 + 16, &Bs[brow][bcol + 16]);
        }
        if (k0 + 64 < K) {
            __builtin_prefetch(gA + k0 + 64, 0, 3);
            if (tid < 128) __builtin_prefetch(gB + k0 + 64, 0, 3);
        }
        stage_wait();
        __syncthreads();

        v8i a0 = frag_a(&As[mw][0], lane);
        v8i a1 = frag_a(&As[mw + 16][0], lane);
        v8i b0 = frag_b(&Bs[nw][0], lane);
        v8i b1 = frag_b(&Bs[nw + 16][0], lane);
        acc[0][0] = __builtin_amdgcn_wmma_i32_16x16x64_iu8(true, a0, true, b0, acc[0][0], false, false);
        acc[0][1] = __builtin_amdgcn_wmma_i32_16x16x64_iu8(true, a0, true, b1, acc[0][1], false, false);
        acc[1][0] = __builtin_amdgcn_wmma_i32_16x16x64_iu8(true, a1, true, b0, acc[1][0], false, false);
        acc[1][1] = __builtin_amdgcn_wmma_i32_16x16x64_iu8(true, a1, true, b1, acc[1][1], false, false);
    }

    int half = lane >> 4;
#pragma unroll
    for (int j = 0; j < 2; ++j) {
        int ocol = n0 + nw + j * 16 + (lane & 15);
        float swo = sw[ocol];
#pragma unroll
        for (int i = 0; i < 2; ++i) {
#pragma unroll
            for (int r = 0; r < 8; ++r) {
                int trow = m0 + mw + i * 16 + r + half * 8;
                out[(size_t)trow * N + ocol] = (float)acc[i][j][r] * sx[trow] * swo;
            }
        }
    }
}

extern "C" void kernel_launch(void* const* d_in, const int* in_sizes, int n_in,
                              void* d_out, int out_size, void* d_ws, size_t ws_size,
                              hipStream_t stream)
{
    const float* x       = (const float*)d_in[0];
    const float* gate_w  = (const float*)d_in[1];
    const float* up_w    = (const float*)d_in[2];
    const float* down_w  = (const float*)d_in[3];
    const float* ug_L    = (const float*)d_in[4];
    const float* ug_R    = (const float*)d_in[5];
    const float* ug_diag = (const float*)d_in[6];
    const float* dn_L    = (const float*)d_in[7];
    const float* dn_R    = (const float*)d_in[8];
    const float* dn_diag = (const float*)d_in[9];
    (void)n_in; (void)out_size; (void)ws_size;

    const int T = in_sizes[0] / H_DIM;

    size_t off = 0;
    auto wsalloc = [&](size_t bytes) -> void* {
        void* p = (char*)d_ws + off;
        off += (bytes + 255) & ~(size_t)255;
        return p;
    };
    float*  inv_scratch = (float*)wsalloc((size_t)4 * 148 * 296 * sizeof(float));
    float*  ugLinvT = (float*)wsalloc((size_t)UG_DL * UG_DL * sizeof(float));
    float*  ugRinvT = (float*)wsalloc((size_t)UG_DR * UG_DR * sizeof(float));
    float*  dnLinvT = (float*)wsalloc((size_t)DN_DL * DN_DL * sizeof(float));
    float*  dnRinvT = (float*)wsalloc((size_t)DN_DR * DN_DR * sizeof(float));
    int8_t* qx_ug   = (int8_t*)wsalloc((size_t)T * H_DIM);
    float*  sx_ug   = (float*)wsalloc((size_t)T * sizeof(float));
    int8_t* qw_up   = (int8_t*)wsalloc((size_t)I_DIM * H_DIM);
    float*  sw_up   = (float*)wsalloc((size_t)I_DIM * sizeof(float));
    int8_t* qw_gate = (int8_t*)wsalloc((size_t)I_DIM * H_DIM);
    float*  sw_gate = (float*)wsalloc((size_t)I_DIM * sizeof(float));
    float*  x_act   = (float*)wsalloc((size_t)T * I_DIM * sizeof(float));
    int8_t* qx_dn   = (int8_t*)wsalloc((size_t)T * I_DIM);
    float*  sx_dn   = (float*)wsalloc((size_t)T * sizeof(float));
    int8_t* qw_dn   = (int8_t*)wsalloc((size_t)H_DIM * I_DIM);
    float*  sw_dn   = (float*)wsalloc((size_t)H_DIM * sizeof(float));

    (void)hipFuncSetAttribute(reinterpret_cast<const void*>(kron_quant),
                              hipFuncAttributeMaxDynamicSharedMemorySize,
                              (int)(2 * DN_DL * DN_DR * sizeof(float) + 2048));

    gauss_inv_t<<<4, 256, 0, stream>>>(ug_L, ug_R, dn_L, dn_R,
                                       ugLinvT, ugRinvT, dnLinvT, dnRinvT,
                                       inv_scratch);

    const size_t ug_lds = (size_t)2 * UG_DL * UG_DR * sizeof(float);
    const size_t dn_lds = (size_t)2 * DN_DL * DN_DR * sizeof(float);

    kron_quant<<<T, 256, ug_lds, stream>>>(x, ug_L, ug_R, ug_diag,
                                           UG_DL, UG_DR, 0, qx_ug, sx_ug);
    kron_quant<<<I_DIM, 256, ug_lds, stream>>>(up_w, ugLinvT, ugRinvT, ug_diag,
                                               UG_DL, UG_DR, 1, qw_up, sw_up);
    kron_quant<<<I_DIM, 256, ug_lds, stream>>>(gate_w, ugLinvT, ugRinvT, ug_diag,
                                               UG_DL, UG_DR, 1, qw_gate, sw_gate);

    gemm_swiglu<<<dim3(I_DIM / 64, T / 128), 256, 0, stream>>>(
        qx_ug, sx_ug, qw_gate, sw_gate, qw_up, sw_up, x_act, T, H_DIM, I_DIM);

    kron_quant<<<T, 256, dn_lds, stream>>>(x_act, dn_L, dn_R, dn_diag,
                                           DN_DL, DN_DR, 0, qx_dn, sx_dn);
    kron_quant<<<H_DIM, 256, dn_lds, stream>>>(down_w, dnLinvT, dnRinvT, dn_diag,
                                               DN_DL, DN_DR, 1, qw_dn, sw_dn);

    gemm_down<<<dim3(H_DIM / 64, T / 128), 256, 0, stream>>>(
        qx_dn, sx_dn, qw_dn, sw_dn, (float*)d_out, T, I_DIM, H_DIM);
}
